// GIN_89318139887641
// MI455X (gfx1250) — matrix operand order repaired and checked by reading, not compiled, not run on hardware
//
#include <hip/hip_runtime.h>
#include <hip/hip_bf16.h>

// ---------------- problem constants (from reference) ----------------
#define NODES   100000
#define EDGES   1600000
#define HDIM    128
#define NGRAPH  64
#define DCAT    768            // 5*H + D_IN
#define NCLASS  10
#define BN_EPS  1e-5f

typedef __bf16 bf16_t;
typedef __attribute__((ext_vector_type(4)))  __bf16 v4bf;
typedef __attribute__((ext_vector_type(8)))  __bf16 v8bf;
typedef __attribute__((ext_vector_type(16))) __bf16 v16bf;
typedef __attribute__((ext_vector_type(8)))  float  v8f;

static __device__ __forceinline__ void atomAddF(float* p, float v) {
  unsafeAtomicAdd(p, v);   // lowers to global_atomic_add_f32 (L2-resident accumulators)
}

// ---------------- tiny utility kernels ----------------
__global__ void k_zero(float* p, int n) {
  int i = blockIdx.x * blockDim.x + threadIdx.x;
  if (i < n) p[i] = 0.0f;
}

// fp32 -> bf16 (x4 vectorized)
__global__ void k_f2bf(const float* __restrict__ in, bf16_t* __restrict__ out, int n4) {
  int i = blockIdx.x * blockDim.x + threadIdx.x;
  if (i >= n4) return;
  float4 f = ((const float4*)in)[i];
  v4bf o;
  o[0] = (bf16_t)f.x; o[1] = (bf16_t)f.y; o[2] = (bf16_t)f.z; o[3] = (bf16_t)f.w;
  ((v4bf*)out)[i] = o;
}

// bf16 -> fp32 (x4) : initializes agg with the self term (x + sum_neighbors)
__global__ void k_bf2f(const bf16_t* __restrict__ in, float* __restrict__ out, int n4) {
  int i = blockIdx.x * blockDim.x + threadIdx.x;
  if (i >= n4) return;
  v4bf b = ((const v4bf*)in)[i];
  float4 f;
  f.x = (float)b[0]; f.y = (float)b[1]; f.z = (float)b[2]; f.w = (float)b[3];
  ((float4*)out)[i] = f;
}

// transpose 128x128 fp32 weight (k-major) -> bf16 [n][k] so B-operand K is contiguous
__global__ void k_wt(const float* __restrict__ W, bf16_t* __restrict__ Wt) {
  int idx = blockIdx.x * blockDim.x + threadIdx.x;   // 16384
  int k = idx >> 7, n = idx & 127;
  Wt[n * HDIM + k] = (bf16_t)W[idx];
}

// ---------------- edge scatter-add (GIN aggregation) ----------------
// One thread handles 8 features of one edge: bf16 gather, fp32 atomic accumulate.
__global__ void k_scatter(const bf16_t* __restrict__ feat,
                          const int* __restrict__ src, const int* __restrict__ dst,
                          float* __restrict__ agg) {
  int tid = blockIdx.x * blockDim.x + threadIdx.x;
  const int total = EDGES * 16;
  if (tid >= total) return;
  int e  = tid >> 4;
  int c0 = (tid & 15) << 3;
  int s = src[e], d = dst[e];
  v8bf v = *(const v8bf*)(feat + (size_t)s * HDIM + c0);
  float* base = agg + (size_t)d * HDIM + c0;
#pragma unroll
  for (int i = 0; i < 8; ++i) atomAddF(base + i, (float)v[i]);
}

// ---------------- WMMA bf16 GEMM: out[N,128] = A[N,128] @ W + bias ----------------
// Wt is W transposed ([n][k], bf16) so both operands read contiguous K.
static __device__ __forceinline__ v16bf frag_a(const bf16_t* rowbase, int koff) {
  // 16-bit A 16x32 layout: lanes 0-15 hold K{0..7,16..23}, lanes 16-31 K{8..15,24..31}
  v8bf lo = *(const v8bf*)(rowbase + koff);
  v8bf hi = *(const v8bf*)(rowbase + koff + 16);
  return __builtin_shufflevector(lo, hi, 0,1,2,3,4,5,6,7,8,9,10,11,12,13,14,15);
}
static __device__ __forceinline__ v16bf frag_b(const bf16_t* colbase, int koff) {
  // 16-bit B 32x16 layout: lanes 0-15 hold K{0..7,16..23}, lanes 16-31 K{8..15,24..31}
  v8bf lo = *(const v8bf*)(colbase + koff);
  v8bf hi = *(const v8bf*)(colbase + koff + 16);
  return __builtin_shufflevector(lo, hi, 0,1,2,3,4,5,6,7,8,9,10,11,12,13,14,15);
}

__global__ void __launch_bounds__(256)
k_gemm(const bf16_t* __restrict__ A, const bf16_t* __restrict__ Wt,
       const float* __restrict__ bias, float* __restrict__ out, int nrows) {
  __shared__ __align__(16) bf16_t lds_w[HDIM * HDIM];   // 32 KB of the 320 KB WGP LDS
  // stage W^T into LDS (all 8 waves reuse it 782x across the grid dimension)
  for (int i = threadIdx.x; i < (HDIM * HDIM) / 8; i += blockDim.x)
    ((v8bf*)lds_w)[i] = ((const v8bf*)Wt)[i];
  __syncthreads();

  const int wave = threadIdx.x >> 5;
  const int lane = threadIdx.x & 31;
  const int lr   = lane & 15;
  const int hiG  = lane >> 4;           // 0 for lanes 0-15, 1 for 16-31
  const int rowBase = blockIdx.x * 128 + wave * 16;

  int arow = rowBase + lr;
  if (arow >= nrows) arow = nrows - 1;  // clamp loads; stores are guarded
  const bf16_t* aptr = A + (size_t)arow * HDIM;

  v8f acc[8];
#pragma unroll
  for (int nt = 0; nt < 8; ++nt) acc[nt] = (v8f){0.f,0.f,0.f,0.f,0.f,0.f,0.f,0.f};

#pragma unroll
  for (int ks = 0; ks < 4; ++ks) {              // K = 4 x 32
    const int koffA = ks * 32 + (hiG << 3);     // +0 / +8
    const int koffB = ks * 32 + (hiG << 3);     // +0 / +8
    v16bf af = frag_a(aptr, koffA);
#pragma unroll
    for (int nt = 0; nt < 8; ++nt) {
      v16bf bf = frag_b(lds_w + (nt * 16 + lr) * HDIM, koffB);
      acc[nt] = __builtin_amdgcn_wmma_f32_16x16x32_bf16(
          false, af, false, bf, (short)0, acc[nt], false, false);
    }
  }

  // C/D layout: lanes 0-15 -> rows M=0..7, lanes 16-31 -> rows M=8..15; col = lane&15
  const int mAdd = hiG * 8;
#pragma unroll
  for (int nt = 0; nt < 8; ++nt) {
    const int col = nt * 16 + lr;
    const float bv = bias[col];
#pragma unroll
    for (int r = 0; r < 8; ++r) {
      int m = rowBase + mAdd + r;
      if (m < nrows) out[(size_t)m * HDIM + col] = acc[nt][r] + bv;
    }
  }
}

// ---------------- BatchNorm pieces ----------------
__global__ void k_colstats(const float* __restrict__ h, float* __restrict__ sums) {
  // block: 128 threads (one column each), 256 rows per block
  int c  = threadIdx.x;
  int r0 = blockIdx.x * 256;
  int r1 = r0 + 256; if (r1 > NODES) r1 = NODES;
  float s = 0.f, q = 0.f;
  for (int r = r0; r < r1; ++r) {
    float v = h[(size_t)r * HDIM + c];
    s += v; q += v * v;
  }
  atomAddF(&sums[c], s);
  atomAddF(&sums[HDIM + c], q);
}

__global__ void k_bnfin(const float* __restrict__ sums,
                        const float* __restrict__ gamma, const float* __restrict__ beta,
                        float* __restrict__ ss) {
  int c = threadIdx.x;                      // 128 threads, 1 block
  float mu  = sums[c] * (1.0f / NODES);
  float var = sums[HDIM + c] * (1.0f / NODES) - mu * mu;
  if (var < 0.f) var = 0.f;                 // guard fp roundoff
  float sc = gamma[c] * rsqrtf(var + BN_EPS);
  ss[c]        = sc;
  ss[HDIM + c] = beta[c] - mu * sc;
}

__global__ void k_bnapply(const float* __restrict__ h, const float* __restrict__ ss,
                          bf16_t* __restrict__ out, int n4) {
  int i = blockIdx.x * blockDim.x + threadIdx.x;
  if (i >= n4) return;
  int c0 = (i * 4) & (HDIM - 1);
  float4 f = ((const float4*)h)[i];
  v4bf o;
  float v0 = fmaxf(f.x * ss[c0 + 0] + ss[HDIM + c0 + 0], 0.f);
  float v1 = fmaxf(f.y * ss[c0 + 1] + ss[HDIM + c0 + 1], 0.f);
  float v2 = fmaxf(f.z * ss[c0 + 2] + ss[HDIM + c0 + 2], 0.f);
  float v3 = fmaxf(f.w * ss[c0 + 3] + ss[HDIM + c0 + 3], 0.f);
  o[0] = (bf16_t)v0; o[1] = (bf16_t)v1; o[2] = (bf16_t)v2; o[3] = (bf16_t)v3;
  ((v4bf*)out)[i] = o;
}

// last BN of each layer: also accumulate per-graph mean-pool partial sums (fp32)
__global__ void k_bnapply_pool(const float* __restrict__ h, const float* __restrict__ ss,
                               bf16_t* __restrict__ out, const int* __restrict__ batch,
                               float* __restrict__ pool, int colOff, int n4) {
  int i = blockIdx.x * blockDim.x + threadIdx.x;
  if (i >= n4) return;
  int base = i * 4;
  int c0 = base & (HDIM - 1);
  int node = base >> 7;
  int g = batch[node];
  float* pb = pool + (size_t)g * DCAT + colOff + c0;
  float4 f = ((const float4*)h)[i];
  float v0 = fmaxf(f.x * ss[c0 + 0] + ss[HDIM + c0 + 0], 0.f);
  float v1 = fmaxf(f.y * ss[c0 + 1] + ss[HDIM + c0 + 1], 0.f);
  float v2 = fmaxf(f.z * ss[c0 + 2] + ss[HDIM + c0 + 2], 0.f);
  float v3 = fmaxf(f.w * ss[c0 + 3] + ss[HDIM + c0 + 3], 0.f);
  v4bf o;
  o[0] = (bf16_t)v0; o[1] = (bf16_t)v1; o[2] = (bf16_t)v2; o[3] = (bf16_t)v3;
  ((v4bf*)out)[i] = o;
  atomAddF(pb + 0, v0); atomAddF(pb + 1, v1);
  atomAddF(pb + 2, v2); atomAddF(pb + 3, v3);
}

// ---------------- pooling / head ----------------
__global__ void k_poolx(const float* __restrict__ x, const int* __restrict__ batch,
                        float* __restrict__ pool, int n4) {
  int i = blockIdx.x * blockDim.x + threadIdx.x;
  if (i >= n4) return;
  int base = i * 4;
  int c0 = base & (HDIM - 1);
  int node = base >> 7;
  int g = batch[node];
  float4 f = ((const float4*)x)[i];
  float* pb = pool + (size_t)g * DCAT + c0;
  atomAddF(pb + 0, f.x); atomAddF(pb + 1, f.y);
  atomAddF(pb + 2, f.z); atomAddF(pb + 3, f.w);
}

__global__ void k_count(const int* __restrict__ batch, float* __restrict__ cnt) {
  int n = blockIdx.x * blockDim.x + threadIdx.x;
  if (n < NODES) atomAddF(&cnt[batch[n]], 1.0f);
}

__global__ void k_mean(float* __restrict__ pool, const float* __restrict__ cnt) {
  int i = blockIdx.x * blockDim.x + threadIdx.x;
  if (i >= NGRAPH * DCAT) return;
  pool[i] /= fmaxf(cnt[i / DCAT], 1.0f);
}

__global__ void k_fc1(const float* __restrict__ hg, const float* __restrict__ W,
                      const float* __restrict__ b, float* __restrict__ out) {
  int o = blockIdx.x * blockDim.x + threadIdx.x;   // 64*768
  if (o >= NGRAPH * DCAT) return;
  int g = o / DCAT, j = o - g * DCAT;
  const float* hrow = hg + (size_t)g * DCAT;
  float s = b[j];
  for (int k = 0; k < DCAT; ++k) s += hrow[k] * W[(size_t)k * DCAT + j];
  out[o] = fmaxf(s, 0.f);
}

__global__ void k_fc2(const float* __restrict__ hg2, const float* __restrict__ W,
                      const float* __restrict__ b, float* __restrict__ out) {
  int o = blockIdx.x * blockDim.x + threadIdx.x;   // 64*10
  if (o >= NGRAPH * NCLASS) return;
  int g = o / NCLASS, j = o - g * NCLASS;
  const float* hrow = hg2 + (size_t)g * DCAT;
  float s = b[j];
  for (int k = 0; k < DCAT; ++k) s += hrow[k] * W[(size_t)k * NCLASS + j];
  out[o] = s;
}

// ---------------- host orchestration ----------------
extern "C" void kernel_launch(void* const* d_in, const int* in_sizes, int n_in,
                              void* d_out, int out_size, void* d_ws, size_t ws_size,
                              hipStream_t stream) {
  (void)in_sizes; (void)n_in; (void)out_size; (void)ws_size;

  const float* x   = (const float*)d_in[0];
  const int*   src = (const int*)d_in[1];
  const int*   dst = src + EDGES;
  const int*   bat = (const int*)d_in[2];
  // layer i params: base 3 + 8i : W1,b1,g1,be1,W2,b2,gn,bn ; fc at 43..46
  const float* fcW1 = (const float*)d_in[43];
  const float* fcB1 = (const float*)d_in[44];
  const float* fcW3 = (const float*)d_in[45];
  const float* fcB3 = (const float*)d_in[46];
  float* out = (float*)d_out;

  // workspace carve-up (256B aligned)
  char* ws = (char*)d_ws;
  size_t off = 0;
  auto carve = [&](size_t bytes) { char* p = ws + off; off = (off + bytes + 255) & ~(size_t)255; return p; };
  const size_t NH = (size_t)NODES * HDIM;
  bf16_t* featA = (bf16_t*)carve(NH * 2);
  bf16_t* featB = (bf16_t*)carve(NH * 2);
  bf16_t* abuf  = (bf16_t*)carve(NH * 2);
  float*  agg   = (float*)carve(NH * 4);
  float*  htmp  = (float*)carve(NH * 4);
  bf16_t* wt    = (bf16_t*)carve((size_t)10 * HDIM * HDIM * 2);
  float*  stats = (float*)carve(2 * HDIM * 4);   // colsum, colsumsq
  float*  ss    = (float*)carve(2 * HDIM * 4);   // scale, shift
  float*  pool  = (float*)carve((size_t)NGRAPH * DCAT * 4);
  float*  cnt   = (float*)carve(NGRAPH * 4);
  float*  hg2   = (float*)carve((size_t)NGRAPH * DCAT * 4);

  const int n4      = (int)(NH / 4);
  const int gN4     = (n4 + 255) / 256;
  const int gScat   = (EDGES * 16 + 255) / 256;
  const int gGemm   = (NODES + 127) / 128;
  const int gStats  = (NODES + 255) / 256;

  // ---- one-time per call prep ----
  for (int i = 0; i < 5; ++i) {
    const float* W1 = (const float*)d_in[3 + 8 * i + 0];
    const float* W2 = (const float*)d_in[3 + 8 * i + 4];
    k_wt<<<64, 256, 0, stream>>>(W1, wt + (size_t)(2 * i + 0) * HDIM * HDIM);
    k_wt<<<64, 256, 0, stream>>>(W2, wt + (size_t)(2 * i + 1) * HDIM * HDIM);
  }
  k_f2bf<<<gN4, 256, 0, stream>>>(x, featA, n4);
  k_zero<<<(NGRAPH * DCAT + 255) / 256, 256, 0, stream>>>(pool, NGRAPH * DCAT);
  k_zero<<<1, 256, 0, stream>>>(cnt, NGRAPH);
  k_poolx<<<gN4, 256, 0, stream>>>(x, bat, pool, n4);
  k_count<<<(NODES + 255) / 256, 256, 0, stream>>>(bat, cnt);

  // ---- 5 GIN layers ----
  bf16_t* cur = featA;
  bf16_t* nxt = featB;
  for (int i = 0; i < 5; ++i) {
    const float* b1  = (const float*)d_in[3 + 8 * i + 1];
    const float* g1  = (const float*)d_in[3 + 8 * i + 2];
    const float* be1 = (const float*)d_in[3 + 8 * i + 3];
    const float* b2  = (const float*)d_in[3 + 8 * i + 5];
    const float* gn  = (const float*)d_in[3 + 8 * i + 6];
    const float* bn  = (const float*)d_in[3 + 8 * i + 7];
    bf16_t* Wt1 = wt + (size_t)(2 * i + 0) * HDIM * HDIM;
    bf16_t* Wt2 = wt + (size_t)(2 * i + 1) * HDIM * HDIM;

    // agg = x + sum_{j in N(i)} x_j  (fp32 accumulators, L2-resident)
    k_bf2f<<<gN4, 256, 0, stream>>>(cur, agg, n4);
    k_scatter<<<gScat, 256, 0, stream>>>(cur, src, dst, agg);
    k_f2bf<<<gN4, 256, 0, stream>>>(agg, abuf, n4);

    // h = (x+agg) @ W1 + b1 ; BN ; ReLU
    k_gemm<<<gGemm, 256, 0, stream>>>(abuf, Wt1, b1, htmp, NODES);
    k_zero<<<1, 256, 0, stream>>>(stats, 2 * HDIM);
    k_colstats<<<gStats, 128, 0, stream>>>(htmp, stats);
    k_bnfin<<<1, 128, 0, stream>>>(stats, g1, be1, ss);
    k_bnapply<<<gN4, 256, 0, stream>>>(htmp, ss, abuf, n4);

    // h = h @ W2 + b2 ; outer BN ; ReLU ; fused mean-pool partials
    k_gemm<<<gGemm, 256, 0, stream>>>(abuf, Wt2, b2, htmp, NODES);
    k_zero<<<1, 256, 0, stream>>>(stats, 2 * HDIM);
    k_colstats<<<gStats, 128, 0, stream>>>(htmp, stats);
    k_bnfin<<<1, 128, 0, stream>>>(stats, gn, bn, ss);
    k_bnapply_pool<<<gN4, 256, 0, stream>>>(htmp, ss, nxt, bat, pool,
                                            HDIM * (i + 1), n4);
    bf16_t* t = cur; cur = nxt; nxt = t;
  }

  // ---- head ----
  k_mean<<<(NGRAPH * DCAT + 255) / 256, 256, 0, stream>>>(pool, cnt);
  k_fc1<<<(NGRAPH * DCAT + 255) / 256, 256, 0, stream>>>(pool, fcW1, fcB1, hg2);
  k_fc2<<<(NGRAPH * NCLASS + 255) / 256, 256, 0, stream>>>(hg2, fcW3, fcB3, out);
}
